// StructureNetwork_2877628088530
// MI455X (gfx1250) — matrix operand order repaired and checked by reading, not compiled, not run on hardware
//
#include <hip/hip_runtime.h>
#include <hip/hip_bf16.h>

typedef unsigned short u16;
typedef __bf16 v16bf __attribute__((ext_vector_type(16)));
typedef float v8f    __attribute__((ext_vector_type(8)));
typedef unsigned int v4u __attribute__((ext_vector_type(4)));

union Frag  { v4u q[2]; v16bf v; };
union Pack8 { u16 u[8]; v4u q; };

#define NATOM 32768
#define MNBR  12
#define CF    128
#define FNBR  64
#define FIN   92
#define KCONV 320   // 2C + F_nbr
#define OCONV 256
#define KEMB  96    // 92 padded to 3*32
#define KFC   160   // 136 padded to 5*32
#define HDIM  256
#define NCRY  512
#define ROWS_PB 48  // lcm(16 row-tile, 12 nbrs)

__device__ __forceinline__ u16 f2bf(float f) {
  unsigned int u = __float_as_uint(f);
  u += 0x7FFFu + ((u >> 16) & 1u);         // round-to-nearest-even
  return (u16)(u >> 16);
}
__device__ __forceinline__ float softplusf(float v) {
  return (v > 20.f) ? v : log1pf(__expf(v));
}
__device__ __forceinline__ float sigmoidf(float v) {
  return 1.f / (1.f + __expf(-v));
}

// CDNA5 async global->LDS copy (16B per lane), tracked by ASYNCcnt.
__device__ __forceinline__ void async_b128(const void* gaddr, void* ldsp) {
  unsigned lds = (unsigned)(uintptr_t)ldsp;  // low 32 bits of flat LDS addr = LDS offset
  asm volatile("global_load_async_to_lds_b128 %0, %1, off"
               :: "v"(lds), "v"(gaddr) : "memory");
}
__device__ __forceinline__ void wait_async0() {
  asm volatile("s_wait_asynccnt 0x0" ::: "memory");
}

// ---------------- weight prep: transpose + fp32->bf16 (+K padding) ----------
__global__ __launch_bounds__(256) void k_prep_weights(
    const float* __restrict__ conv_W, const float* __restrict__ emb_W,
    const float* __restrict__ fc_W,
    u16* __restrict__ WT, u16* __restrict__ embT, u16* __restrict__ fcT) {
  int idx = blockIdx.x * 256 + threadIdx.x;
  const int nconv = 3 * OCONV * KCONV;   // 245760
  const int nemb  = CF * KEMB;           // 12288
  const int nfc   = HDIM * KFC;          // 40960
  if (idx < nconv) {
    int i = idx / (OCONV * KCONV);
    int rem = idx % (OCONV * KCONV);
    int o = rem / KCONV, k = rem % KCONV;
    WT[idx] = f2bf(conv_W[(size_t)i * KCONV * OCONV + (size_t)k * OCONV + o]);
  } else if (idx < nconv + nemb) {
    int j = idx - nconv;
    int o = j / KEMB, k = j % KEMB;
    embT[j] = (k < FIN) ? f2bf(emb_W[(size_t)k * CF + o]) : (u16)0;
  } else if (idx < nconv + nemb + nfc) {
    int j = idx - nconv - nemb;
    int o = j / KFC, k = j % KFC;
    fcT[j] = (k < CF + 8) ? f2bf(fc_W[(size_t)k * HDIM + o]) : (u16)0;
  }
}

// ---------------- one-time nbr_fea fp32 -> bf16 ------------------------------
__global__ __launch_bounds__(256) void k_nbr2bf(
    const float* __restrict__ nbr_fea, u16* __restrict__ nbr_bf) {
  int ch = blockIdx.x * 256 + threadIdx.x;        // 8 elems per thread
  const float* p = nbr_fea + (size_t)ch * 8;
  Pack8 t;
#pragma unroll
  for (int j = 0; j < 8; j++) t.u[j] = f2bf(p[j]);
  *(v4u*)(nbr_bf + (size_t)ch * 8) = t.q;
}

// ---------------- embedding GEMM: x = atom_fea @ emb_W + emb_b --------------
__global__ __launch_bounds__(256) void k_embed(
    const float* __restrict__ atom_fea, const u16* __restrict__ embT,
    const float* __restrict__ emb_b, float* __restrict__ x, u16* __restrict__ xbf) {
  __shared__ u16 Alds[64 * KEMB];
  int tid = threadIdx.x;
  int R = blockIdx.x * 64;
  for (int i = tid; i < 64 * KEMB; i += 256) {
    int rl = i / KEMB, c = i % KEMB;
    float v = (c < FIN) ? atom_fea[(size_t)(R + rl) * FIN + c] : 0.f;
    Alds[i] = f2bf(v);
  }
  __syncthreads();
  int wid = tid >> 5, lane = tid & 31, l15 = lane & 15, hi = lane >> 4;
  int ksel = hi ? 8 : 0, kba = hi ? 8 : 0;
  int col = wid * 16 + l15;
  v8f z0 = {0.f, 0.f, 0.f, 0.f, 0.f, 0.f, 0.f, 0.f};
  v8f acc[4];
#pragma unroll
  for (int rt = 0; rt < 4; rt++) acc[rt] = z0;
  for (int ks = 0; ks < KEMB / 32; ks++) {
    int k0 = ks * 32;
    Frag b, a[4];
    const u16* pb = embT + (size_t)col * KEMB + k0 + ksel;
    b.q[0] = *(const v4u*)pb; b.q[1] = *(const v4u*)(pb + 16);
#pragma unroll
    for (int rt = 0; rt < 4; rt++) {
      const u16* pa = Alds + (rt * 16 + l15) * KEMB + k0 + kba;
      a[rt].q[0] = *(const v4u*)pa; a[rt].q[1] = *(const v4u*)(pa + 16);
    }
#pragma unroll
    for (int rt = 0; rt < 4; rt++)
      acc[rt] = __builtin_amdgcn_wmma_f32_16x16x32_bf16(
          false, a[rt].v, false, b.v, (short)0, acc[rt], false, false);
  }
  float bb = emb_b[col];
#pragma unroll
  for (int rt = 0; rt < 4; rt++) {
#pragma unroll
    for (int v = 0; v < 8; v++) {
      int row = R + rt * 16 + v + hi * 8;
      float y = acc[rt][v] + bb;
      x[(size_t)row * CF + col] = y;
      xbf[(size_t)row * CF + col] = f2bf(y);
    }
  }
}

// ---------------- conv helpers ----------------------------------------------
// Stage A tile: 48 rows of tot = [x[n] | x[nbr_idx[n,m]] | nbr_fea[n,m]] (bf16).
// x parts (already bf16) go through the async global->LDS path (16B/lane).
template <bool NBR_BF>
__device__ __forceinline__ void conv_stage_A(
    u16* Alds, const u16* __restrict__ xbf, const int* __restrict__ nbr_idx,
    const float* __restrict__ nbr_fea, const u16* __restrict__ nbr_bf,
    int R, int tid) {
  // region [0,256): self + neighbor gather, 48 rows x 32 chunks of 8 halfs
  for (int ch = tid; ch < ROWS_PB * 32; ch += 256) {
    int rl = ch >> 5;
    int cc = (ch & 31) * 8;
    int r = R + rl;
    int n = (cc < CF) ? (r / MNBR) : nbr_idx[r];
    int c = (cc < CF) ? cc : cc - CF;
    async_b128(xbf + (size_t)n * CF + c, Alds + rl * KCONV + cc);
  }
  // region [256,320): nbr_fea, 48 rows x 8 chunks
  if (NBR_BF) {
    for (int ch = tid; ch < ROWS_PB * 8; ch += 256) {
      int rl = ch >> 3;
      int cq = (ch & 7) * 8;
      async_b128(nbr_bf + (size_t)(R + rl) * FNBR + cq,
                 Alds + rl * KCONV + 2 * CF + cq);
    }
  } else {
    for (int ch = tid; ch < ROWS_PB * 8; ch += 256) {
      int rl = ch >> 3;
      int cq = (ch & 7) * 8;
      const float* p = nbr_fea + (size_t)(R + rl) * FNBR + cq;
      Pack8 t;
#pragma unroll
      for (int j = 0; j < 8; j++) t.u[j] = f2bf(p[j]);
      *(v4u*)(Alds + rl * KCONV + 2 * CF + cq) = t.q;
    }
  }
  wait_async0();
}

// Each wave owns column tiles {wid, wid+8}: cols [16w,16w+16) and +128 (filt/core pair)
__device__ __forceinline__ void conv_mma(
    const u16* Alds, const u16* __restrict__ WT, int wid, int lane, v8f acc[3][2]) {
  int l15 = lane & 15, hi = lane >> 4;
  int ksel = hi ? 8 : 0, kba = hi ? 8 : 0;
  int col0 = wid * 16 + l15;
  int col1 = 128 + col0;
  for (int ks = 0; ks < KCONV / 32; ks++) {
    int k0 = ks * 32;
    Frag b0, b1, a[3];
    const u16* pb0 = WT + (size_t)col0 * KCONV + k0 + ksel;
    const u16* pb1 = WT + (size_t)col1 * KCONV + k0 + ksel;
    b0.q[0] = *(const v4u*)pb0; b0.q[1] = *(const v4u*)(pb0 + 16);
    b1.q[0] = *(const v4u*)pb1; b1.q[1] = *(const v4u*)(pb1 + 16);
#pragma unroll
    for (int rt = 0; rt < 3; rt++) {
      const u16* pa = Alds + (rt * 16 + l15) * KCONV + k0 + kba;
      a[rt].q[0] = *(const v4u*)pa; a[rt].q[1] = *(const v4u*)(pa + 16);
    }
#pragma unroll
    for (int rt = 0; rt < 3; rt++) {
      acc[rt][0] = __builtin_amdgcn_wmma_f32_16x16x32_bf16(
          false, a[rt].v, false, b0.v, (short)0, acc[rt][0], false, false);
      acc[rt][1] = __builtin_amdgcn_wmma_f32_16x16x32_bf16(
          false, a[rt].v, false, b1.v, (short)0, acc[rt][1], false, false);
    }
  }
}

// ---------------- conv pass 1: GEMM -> BN1 stats (sum, sumsq per column) ----
template <bool NBR_BF>
__global__ __launch_bounds__(256) void k_conv_pass1(
    const u16* __restrict__ xbf, const int* __restrict__ nbr_idx,
    const float* __restrict__ nbr_fea, const u16* __restrict__ nbr_bf,
    const u16* __restrict__ WT, const float* __restrict__ bias,
    float* __restrict__ gstats1) {
  __shared__ u16 Alds[ROWS_PB * KCONV];
  __shared__ float ls[512];
  int tid = threadIdx.x;
  for (int i = tid; i < 512; i += 256) ls[i] = 0.f;
  conv_stage_A<NBR_BF>(Alds, xbf, nbr_idx, nbr_fea, nbr_bf, blockIdx.x * ROWS_PB, tid);
  __syncthreads();
  int wid = tid >> 5, lane = tid & 31, l15 = lane & 15;
  v8f z0 = {0.f, 0.f, 0.f, 0.f, 0.f, 0.f, 0.f, 0.f};
  v8f acc[3][2];
#pragma unroll
  for (int rt = 0; rt < 3; rt++) { acc[rt][0] = z0; acc[rt][1] = z0; }
  conv_mma(Alds, WT, wid, lane, acc);
#pragma unroll
  for (int cc = 0; cc < 2; cc++) {
    int col = cc * 128 + wid * 16 + l15;
    float b = bias[col];
    float s = 0.f, sq = 0.f;
#pragma unroll
    for (int rt = 0; rt < 3; rt++) {
#pragma unroll
      for (int v = 0; v < 8; v++) {
        float z = acc[rt][cc][v] + b;
        s += z; sq += z * z;
      }
    }
    atomicAdd(&ls[col], s);
    atomicAdd(&ls[256 + col], sq);
  }
  __syncthreads();
  for (int i = tid; i < 512; i += 256) atomicAdd(&gstats1[i], ls[i]);
}

__global__ void k_finalize1(const float* __restrict__ gstats1,
                            const float* __restrict__ g1, const float* __restrict__ b1,
                            float* __restrict__ ss1) {
  int c = threadIdx.x;  // 256 threads
  float cnt = (float)(NATOM * MNBR);
  float mu = gstats1[c] / cnt;
  float var = gstats1[256 + c] / cnt - mu * mu;
  float sc = g1[c] * rsqrtf(var + 1e-5f);
  ss1[c] = sc;
  ss1[256 + c] = b1[c] - mu * sc;
}

// ---------------- conv pass 2: GEMM -> BN1 affine -> sig*softplus -> sum_m --
template <bool NBR_BF>
__global__ __launch_bounds__(256) void k_conv_pass2(
    const u16* __restrict__ xbf, const int* __restrict__ nbr_idx,
    const float* __restrict__ nbr_fea, const u16* __restrict__ nbr_bf,
    const u16* __restrict__ WT, const float* __restrict__ bias,
    const float* __restrict__ ss1,
    float* __restrict__ summed, float* __restrict__ gstats2) {
  __shared__ u16 Alds[ROWS_PB * KCONV];
  __shared__ float actbuf[ROWS_PB * CF];
  __shared__ float ls[256];
  int tid = threadIdx.x;
  for (int i = tid; i < 256; i += 256) ls[i] = 0.f;
  int R = blockIdx.x * ROWS_PB;
  conv_stage_A<NBR_BF>(Alds, xbf, nbr_idx, nbr_fea, nbr_bf, R, tid);
  __syncthreads();
  int wid = tid >> 5, lane = tid & 31, l15 = lane & 15, hi = lane >> 4;
  v8f z0 = {0.f, 0.f, 0.f, 0.f, 0.f, 0.f, 0.f, 0.f};
  v8f acc[3][2];
#pragma unroll
  for (int rt = 0; rt < 3; rt++) { acc[rt][0] = z0; acc[rt][1] = z0; }
  conv_mma(Alds, WT, wid, lane, acc);
  int colf = wid * 16 + l15, colc = 128 + colf;
  float scf = ss1[colf], shf = ss1[256 + colf], bfv = bias[colf];
  float scc = ss1[colc], shc = ss1[256 + colc], bcv = bias[colc];
#pragma unroll
  for (int rt = 0; rt < 3; rt++) {
#pragma unroll
    for (int v = 0; v < 8; v++) {
      int row_l = rt * 16 + v + hi * 8;
      float zf = (acc[rt][0][v] + bfv) * scf + shf;
      float zc = (acc[rt][1][v] + bcv) * scc + shc;
      actbuf[row_l * CF + colf] = sigmoidf(zf) * softplusf(zc);
    }
  }
  __syncthreads();
  int abase = R / MNBR;
  for (int it = tid; it < (ROWS_PB / MNBR) * CF; it += 256) {  // 4 atoms x 128
    int a = it / CF, c = it % CF;
    const float* pb = actbuf + (a * MNBR) * CF + c;
    float s = 0.f;
#pragma unroll
    for (int m = 0; m < MNBR; m++) s += pb[m * CF];
    summed[(size_t)(abase + a) * CF + c] = s;
    atomicAdd(&ls[c], s);
    atomicAdd(&ls[128 + c], s * s);
  }
  __syncthreads();
  for (int i = tid; i < 256; i += 256) atomicAdd(&gstats2[i], ls[i]);
}

__global__ void k_finalize2(const float* __restrict__ gstats2,
                            const float* __restrict__ g2, const float* __restrict__ b2,
                            float* __restrict__ ss2) {
  int c = threadIdx.x;  // 128 threads
  float cnt = (float)NATOM;
  float mu = gstats2[c] / cnt;
  float var = gstats2[128 + c] / cnt - mu * mu;
  float sc = g2[c] * rsqrtf(var + 1e-5f);
  ss2[c] = sc;
  ss2[128 + c] = b2[c] - mu * sc;
}

// ---------------- residual + softplus, refresh bf16 copy --------------------
__global__ __launch_bounds__(256) void k_update(
    float* __restrict__ x, u16* __restrict__ xbf,
    const float* __restrict__ summed, const float* __restrict__ ss2) {
  int idx = blockIdx.x * 256 + threadIdx.x;  // exact N*C coverage
  int c = idx & (CF - 1);
  float v = x[idx] + ss2[c] * summed[idx] + ss2[128 + c];
  float y = softplusf(v);
  x[idx] = y;
  xbf[idx] = f2bf(y);
}

// ---------------- segment pooling -------------------------------------------
__global__ __launch_bounds__(256) void k_segsum(
    const float* __restrict__ x, const int* __restrict__ seg,
    float* __restrict__ segsum, float* __restrict__ segcnt) {
  int idx = blockIdx.x * 256 + threadIdx.x;
  int n = idx >> 7, c = idx & 127;
  int s = seg[n];
  atomicAdd(&segsum[(size_t)s * CF + c], x[idx]);
  if (c == 0) atomicAdd(&segcnt[s], 1.0f);
}

__global__ __launch_bounds__(256) void k_fcin(
    const float* __restrict__ segsum, const float* __restrict__ segcnt,
    const float* __restrict__ cell_fea, u16* __restrict__ fcin) {
  int idx = blockIdx.x * 256 + threadIdx.x;
  if (idx >= NCRY * KFC) return;
  int n0 = idx / KFC, k = idx % KFC;
  float v;
  if (k < CF) {
    float cnt = fmaxf(segcnt[n0], 1.f);
    v = softplusf(segsum[(size_t)n0 * CF + k] / cnt);
  } else if (k < CF + 8) {
    v = cell_fea[n0 * 8 + (k - CF)];
  } else {
    v = 0.f;
  }
  fcin[idx] = f2bf(v);
}

// ---------------- final FC GEMM + softplus ----------------------------------
__global__ __launch_bounds__(256) void k_fc(
    const u16* __restrict__ fcin, const u16* __restrict__ fcT,
    const float* __restrict__ fc_b, float* __restrict__ out) {
  __shared__ u16 Alds[16 * KFC];
  int tid = threadIdx.x;
  int R = blockIdx.x * 16;
  for (int i = tid; i < 16 * KFC / 8; i += 256)
    async_b128(fcin + (size_t)R * KFC + i * 8, Alds + i * 8);
  wait_async0();
  __syncthreads();
  int wid = tid >> 5, lane = tid & 31, l15 = lane & 15, hi = lane >> 4;
  int ksel = hi ? 8 : 0, kba = hi ? 8 : 0;
  int col0 = wid * 16 + l15, col1 = 128 + col0;
  v8f a0 = {0.f, 0.f, 0.f, 0.f, 0.f, 0.f, 0.f, 0.f};
  v8f a1 = a0;
  for (int ks = 0; ks < KFC / 32; ks++) {
    int k0 = ks * 32;
    Frag b0, b1, a;
    const u16* pb0 = fcT + (size_t)col0 * KFC + k0 + ksel;
    const u16* pb1 = fcT + (size_t)col1 * KFC + k0 + ksel;
    b0.q[0] = *(const v4u*)pb0; b0.q[1] = *(const v4u*)(pb0 + 16);
    b1.q[0] = *(const v4u*)pb1; b1.q[1] = *(const v4u*)(pb1 + 16);
    const u16* pa = Alds + l15 * KFC + k0 + kba;
    a.q[0] = *(const v4u*)pa; a.q[1] = *(const v4u*)(pa + 16);
    a0 = __builtin_amdgcn_wmma_f32_16x16x32_bf16(false, a.v, false, b0.v, (short)0, a0, false, false);
    a1 = __builtin_amdgcn_wmma_f32_16x16x32_bf16(false, a.v, false, b1.v, (short)0, a1, false, false);
  }
  float bb0 = fc_b[col0], bb1 = fc_b[col1];
#pragma unroll
  for (int v = 0; v < 8; v++) {
    int row = R + v + hi * 8;
    out[(size_t)row * HDIM + col0] = softplusf(a0[v] + bb0);
    out[(size_t)row * HDIM + col1] = softplusf(a1[v] + bb1);
  }
}

// ---------------- host orchestration ----------------------------------------
extern "C" void kernel_launch(void* const* d_in, const int* in_sizes, int n_in,
                              void* d_out, int out_size, void* d_ws, size_t ws_size,
                              hipStream_t stream) {
  (void)in_sizes; (void)n_in; (void)out_size;
  const float* atom_fea = (const float*)d_in[0];
  const float* nbr_fea  = (const float*)d_in[1];
  const int*   nbr_idx  = (const int*)d_in[2];
  const float* cell_fea = (const float*)d_in[3];
  const int*   seg      = (const int*)d_in[4];
  const float* emb_W    = (const float*)d_in[5];
  const float* emb_b    = (const float*)d_in[6];
  const float* conv_W   = (const float*)d_in[7];
  const float* conv_b   = (const float*)d_in[8];
  const float* bn1_g    = (const float*)d_in[9];
  const float* bn1_b    = (const float*)d_in[10];
  const float* bn2_g    = (const float*)d_in[11];
  const float* bn2_b    = (const float*)d_in[12];
  const float* fc_W     = (const float*)d_in[13];
  const float* fc_b     = (const float*)d_in[14];
  float* out = (float*)d_out;

  char* w = (char*)d_ws;
  auto alloc = [&](size_t bytes) -> void* {
    void* p = (void*)w;
    w += (bytes + 255) & ~(size_t)255;
    return p;
  };
  float* x       = (float*)alloc((size_t)NATOM * CF * 4);
  u16*   xbf     = (u16*)  alloc((size_t)NATOM * CF * 2);
  float* summed  = (float*)alloc((size_t)NATOM * CF * 4);
  u16*   WT      = (u16*)  alloc((size_t)3 * OCONV * KCONV * 2);
  u16*   embT    = (u16*)  alloc((size_t)CF * KEMB * 2);
  u16*   fcT     = (u16*)  alloc((size_t)HDIM * KFC * 2);
  float* gstats1 = (float*)alloc(512 * 4);
  float* ss1     = (float*)alloc(512 * 4);
  float* gstats2 = (float*)alloc(256 * 4);
  float* ss2     = (float*)alloc(256 * 4);
  float* segsum  = (float*)alloc((size_t)NCRY * CF * 4);
  float* segcnt  = (float*)alloc(NCRY * 4);
  u16*   fcin    = (u16*)  alloc((size_t)NCRY * KFC * 2);

  // Optional bf16 copy of nbr_fea (halves dominant HBM traffic) if ws allows.
  size_t used = (size_t)(w - (char*)d_ws);
  size_t nbrbf_bytes = (size_t)NATOM * MNBR * FNBR * 2;
  bool use_nbrbf = (used + nbrbf_bytes + 512) <= ws_size;
  u16* nbrbf = use_nbrbf ? (u16*)alloc(nbrbf_bytes) : (u16*)nullptr;

  k_prep_weights<<<(3 * OCONV * KCONV + CF * KEMB + HDIM * KFC) / 256, 256, 0, stream>>>(
      conv_W, emb_W, fc_W, WT, embT, fcT);
  if (use_nbrbf)
    k_nbr2bf<<<NATOM * MNBR * FNBR / 8 / 256, 256, 0, stream>>>(nbr_fea, nbrbf);
  k_embed<<<NATOM / 64, 256, 0, stream>>>(atom_fea, embT, emb_b, x, xbf);

  const int nblk = NATOM * MNBR / ROWS_PB;  // 8192
  for (int i = 0; i < 3; i++) {
    const u16* WTi = WT + (size_t)i * OCONV * KCONV;
    hipMemsetAsync(gstats1, 0, 512 * 4, stream);
    if (use_nbrbf)
      k_conv_pass1<true><<<nblk, 256, 0, stream>>>(xbf, nbr_idx, nbr_fea, nbrbf,
                                                   WTi, conv_b + i * OCONV, gstats1);
    else
      k_conv_pass1<false><<<nblk, 256, 0, stream>>>(xbf, nbr_idx, nbr_fea, nbrbf,
                                                    WTi, conv_b + i * OCONV, gstats1);
    k_finalize1<<<1, 256, 0, stream>>>(gstats1, bn1_g + i * 256, bn1_b + i * 256, ss1);
    hipMemsetAsync(gstats2, 0, 256 * 4, stream);
    if (use_nbrbf)
      k_conv_pass2<true><<<nblk, 256, 0, stream>>>(xbf, nbr_idx, nbr_fea, nbrbf, WTi,
                                                   conv_b + i * OCONV, ss1, summed, gstats2);
    else
      k_conv_pass2<false><<<nblk, 256, 0, stream>>>(xbf, nbr_idx, nbr_fea, nbrbf, WTi,
                                                    conv_b + i * OCONV, ss1, summed, gstats2);
    k_finalize2<<<1, 128, 0, stream>>>(gstats2, bn2_g + i * 128, bn2_b + i * 128, ss2);
    k_update<<<NATOM * CF / 256, 256, 0, stream>>>(x, xbf, summed, ss2);
  }

  hipMemsetAsync(segsum, 0, (size_t)NCRY * CF * 4, stream);
  hipMemsetAsync(segcnt, 0, NCRY * 4, stream);
  k_segsum<<<NATOM * CF / 256, 256, 0, stream>>>(x, seg, segsum, segcnt);
  k_fcin<<<(NCRY * KFC + 255) / 256, 256, 0, stream>>>(segsum, segcnt, cell_fea, fcin);
  k_fc<<<NCRY / 16, 256, 0, stream>>>(fcin, fcT, fc_b, out);
}
